// GraphFuse_90726889161220
// MI455X (gfx1250) — matrix operand order repaired and checked by reading, not compiled, not run on hardware
//
#include <hip/hip_runtime.h>
#include <hip/hip_bf16.h>

#define N_NODES 50000
#define N_EDGES 800000
#define F_DIM 256
#define H_DIM 256
#define O_DIM 64

typedef __bf16 bf16_t;
typedef __attribute__((ext_vector_type(16))) __bf16 v16bf;
typedef __attribute__((ext_vector_type(8)))  __bf16 v8bf;
typedef __attribute__((ext_vector_type(8)))  float  v8f;

__device__ __forceinline__ unsigned short f32_to_bf16_raw(float f) {
    unsigned u = __float_as_uint(f);
    u += 0x7FFFu + ((u >> 16) & 1u);          // round-to-nearest-even
    return (unsigned short)(u >> 16);
}

// ---------------- conversion kernels ----------------

__global__ void cvt_f32_to_bf16(const float* __restrict__ in,
                                unsigned short* __restrict__ out, long long n) {
    long long i = blockIdx.x * (long long)blockDim.x + threadIdx.x;
    long long stride = (long long)gridDim.x * blockDim.x;
    for (; i < n; i += stride) out[i] = f32_to_bf16_raw(in[i]);
}

// Build column-major bf16 weight buffers:
//   wht[c*256+k] = hidden_weight[k*256+c]                (256 cols x 256 k)
//   w2t[c*256+k] = (c<64 ? mean_w[k*64+c] : log_std_w[k*64+c-64])  (128 cols)
__global__ void prep_weights(const float* __restrict__ hw,
                             const float* __restrict__ mw,
                             const float* __restrict__ lw,
                             unsigned short* __restrict__ wht,
                             unsigned short* __restrict__ w2t) {
    int tid = blockIdx.x * blockDim.x + threadIdx.x;
    if (tid < 256 * 256) {
        int c = tid >> 8, k = tid & 255;
        wht[c * 256 + k] = f32_to_bf16_raw(hw[k * 256 + c]);
    }
    int t2 = tid - 256 * 256;
    if (t2 >= 0 && t2 < 128 * 256) {
        int c = t2 >> 8, k = t2 & 255;
        float v = (c < 64) ? mw[k * 64 + c] : lw[k * 64 + (c - 64)];
        w2t[c * 256 + k] = f32_to_bf16_raw(v);
    }
}

// ---------------- WMMA GEMM ----------------
// C[M, Ncols] (f32, row-major) = A[M, K] (bf16 row-major) @ Bt[Ncols, K] (bf16, col-major weights)
// One wave (32 threads) per block computes a 16 x 64 tile.
// grid = (M/16, Ncols/64). K multiple of 32.
__global__ __launch_bounds__(32)
void gemm_bf16_wmma(const unsigned short* __restrict__ A,
                    const unsigned short* __restrict__ Bt,
                    float* __restrict__ C,
                    int M, int Ncols, int K) {
    const int lane  = threadIdx.x;
    const int half  = lane >> 4;      // 0: lanes 0-15, 1: lanes 16-31
    const int l16   = lane & 15;
    const int tile_m = blockIdx.x * 16;
    const int tile_n = blockIdx.y * 64;

    v8f acc[4];
    #pragma unroll
    for (int j = 0; j < 4; ++j)
        #pragma unroll
        for (int i = 0; i < 8; ++i) acc[j][i] = 0.0f;

    const bf16_t* Ab = (const bf16_t*)A;
    const bf16_t* Bb = (const bf16_t*)Bt;
    const long long arow_base = (long long)(tile_m + l16) * K;

    for (int kk = 0; kk < K; kk += 32) {
        // A fragment: lanes 0-15 -> K {0..7, 16..23}; lanes 16-31 -> K {8..15, 24..31}
        const bf16_t* ap = Ab + arow_base + kk + half * 8;
        v8bf a_lo = *(const v8bf*)(ap);
        v8bf a_hi = *(const v8bf*)(ap + 16);
        v16bf afrag;
        #pragma unroll
        for (int i = 0; i < 8; ++i) { afrag[i] = a_lo[i]; afrag[i + 8] = a_hi[i]; }

        #pragma unroll
        for (int j = 0; j < 4; ++j) {
            // B fragment: lanes 0-15 -> K {0..7, 16..23}; lanes 16-31 -> K {8..15, 24..31}
            const int col = tile_n + j * 16 + l16;
            const bf16_t* bp = Bb + (long long)col * K + kk + half * 8;
            v8bf b_lo = *(const v8bf*)(bp);
            v8bf b_hi = *(const v8bf*)(bp + 16);
            v16bf bfrag;
            #pragma unroll
            for (int i = 0; i < 8; ++i) { bfrag[i] = b_lo[i]; bfrag[i + 8] = b_hi[i]; }

            acc[j] = __builtin_amdgcn_wmma_f32_16x16x32_bf16(
                false, afrag, false, bfrag, (short)0, acc[j], false, false);
        }
    }

    // C/D layout: VGPR v -> row (tile_m + v + half*8), col = tile_n + j*16 + l16
    #pragma unroll
    for (int j = 0; j < 4; ++j) {
        const int col = tile_n + j * 16 + l16;
        #pragma unroll
        for (int v = 0; v < 8; ++v) {
            const int row = tile_m + v + half * 8;
            C[(long long)row * Ncols + col] = acc[j][v];
        }
    }
}

// ---------------- SpMM (COO, atomic segment-sum) ----------------
// out[dst[e]*D + :] += w[e] * x[src[e]*D + :]   ; D multiple of 4
__global__ void spmm_atomic(const int* __restrict__ src, const int* __restrict__ dst,
                            const float* __restrict__ w, const float* __restrict__ x,
                            float* __restrict__ out, int E, int D) {
    const int D4 = D >> 2;
    const long long total = (long long)E * D4;
    long long i = blockIdx.x * (long long)blockDim.x + threadIdx.x;
    const long long stride = (long long)gridDim.x * blockDim.x;
    for (; i < total; i += stride) {
        const int e = (int)(i / D4);
        const int c = (int)(i - (long long)e * D4);
        const int s = src[e];
        const int d = dst[e];
        const float ww = w[e];
        const float4 v = ((const float4*)(x + (long long)s * D))[c];
        float* o = out + (long long)d * D + (c << 2);
        atomicAdd(o + 0, ww * v.x);
        atomicAdd(o + 1, ww * v.y);
        atomicAdd(o + 2, ww * v.z);
        atomicAdd(o + 3, ww * v.w);
    }
}

// hidden_g = relu(agg1 + bias) -> bf16 ; hidden_m = relu(support + bias) -> bf16
__global__ void act_kernel(const float* __restrict__ support,
                           const float* __restrict__ agg1,
                           const float* __restrict__ hb,
                           unsigned short* __restrict__ hg,
                           unsigned short* __restrict__ hm, long long n) {
    long long i = blockIdx.x * (long long)blockDim.x + threadIdx.x;
    const long long stride = (long long)gridDim.x * blockDim.x;
    for (; i < n; i += stride) {
        const int h = (int)(i & (H_DIM - 1));
        const float b = hb[h];
        float g = agg1[i] + b;    g = g > 0.f ? g : 0.f;
        float m = support[i] + b; m = m > 0.f ? m : 0.f;
        hg[i] = f32_to_bf16_raw(g);
        hm[i] = f32_to_bf16_raw(m);
    }
}

// mixture: z_mean = gcn*mw + (mlp+bias)*(1-mw) ; z_log_std with sigmoid(mw)
__global__ void mix_kernel(const float* __restrict__ agg2,
                           const float* __restrict__ mlp_out,
                           const float* __restrict__ mean_bias,
                           const float* __restrict__ log_std_bias,
                           const float* __restrict__ mixw,
                           float* __restrict__ out_mean,
                           float* __restrict__ out_ls, int n /* N*O */) {
    int i = blockIdx.x * blockDim.x + threadIdx.x;
    const int stride = gridDim.x * blockDim.x;
    const float mw = mixw[0];
    const float mr = 1.0f / (1.0f + __expf(-mw));
    for (; i < n; i += stride) {
        const int node = i >> 6;
        const int o    = i & 63;
        const long long b = (long long)node * 128;
        const float mg = agg2[b + o];
        const float lg = agg2[b + 64 + o];
        const float mm = mlp_out[b + o]      + mean_bias[o];
        const float lm = mlp_out[b + 64 + o] + log_std_bias[o];
        out_mean[i] = mg * mw + mm * (1.0f - mw);
        out_ls[i]   = lg * mr + lm * (1.0f - mr);
    }
}

extern "C" void kernel_launch(void* const* d_in, const int* in_sizes, int n_in,
                              void* d_out, int out_size, void* d_ws, size_t ws_size,
                              hipStream_t stream) {
    const float* input      = (const float*)d_in[0];
    const int*   edge_index = (const int*)  d_in[1];
    const float* edge_w     = (const float*)d_in[2];
    const float* mix_w      = (const float*)d_in[3];
    const float* hidden_w   = (const float*)d_in[4];
    const float* hidden_b   = (const float*)d_in[5];
    const float* mean_w     = (const float*)d_in[6];
    const float* mean_b     = (const float*)d_in[7];
    const float* log_std_w  = (const float*)d_in[8];
    const float* log_std_b  = (const float*)d_in[9];

    const int* e_src = edge_index;             // edge_index[0]
    const int* e_dst = edge_index + N_EDGES;   // edge_index[1]

    // ---- workspace layout (256B aligned) ----
    char* ws = (char*)d_ws;
    size_t off = 0;
    auto take = [&](size_t bytes) -> char* {
        size_t p = (off + 255) & ~(size_t)255;
        off = p + bytes;
        return ws + p;
    };
    const long long NF  = (long long)N_NODES * F_DIM;   // 12.8M
    const long long NH2 = (long long)N_NODES * 128;     // 6.4M

    unsigned short* in_bf16 = (unsigned short*)take(NF * 2);
    unsigned short* wht     = (unsigned short*)take(256 * 256 * 2);
    unsigned short* w2t     = (unsigned short*)take(128 * 256 * 2);
    float*          support = (float*)take(NF * 4);
    float*          agg1    = (float*)take(NF * 4);
    unsigned short* hg      = (unsigned short*)take(NF * 2);
    unsigned short* hm      = (unsigned short*)take(NF * 2);
    float*          gcn_out = (float*)take(NH2 * 4);
    float*          mlp_out = (float*)take(NH2 * 4);
    float*          agg2    = (float*)take(NH2 * 4);
    (void)ws_size;

    // 1) convert input + weights to bf16 (weights transposed to col-major)
    cvt_f32_to_bf16<<<4096, 256, 0, stream>>>(input, in_bf16, NF);
    prep_weights<<<(256 * 256 + 128 * 256 + 255) / 256, 256, 0, stream>>>(
        hidden_w, mean_w, log_std_w, wht, w2t);

    // 2) support = input @ hidden_weight   [50000 x 256]
    gemm_bf16_wmma<<<dim3(N_NODES / 16, F_DIM / 64), 32, 0, stream>>>(
        in_bf16, wht, support, N_NODES, H_DIM, F_DIM);

    // 3) agg1 = spmm(support)   (zero + atomic scatter)
    hipMemsetAsync(agg1, 0, NF * 4, stream);
    spmm_atomic<<<8192, 256, 0, stream>>>(e_src, e_dst, edge_w, support, agg1,
                                          N_EDGES, H_DIM);

    // 4) hidden_g / hidden_m (relu + bias) -> bf16
    act_kernel<<<4096, 256, 0, stream>>>(support, agg1, hidden_b, hg, hm, NF);

    // 5) gcn_out = hidden_g @ [mean_w | log_std_w]; mlp_out = hidden_m @ same
    gemm_bf16_wmma<<<dim3(N_NODES / 16, 128 / 64), 32, 0, stream>>>(
        hg, w2t, gcn_out, N_NODES, 128, H_DIM);
    gemm_bf16_wmma<<<dim3(N_NODES / 16, 128 / 64), 32, 0, stream>>>(
        hm, w2t, mlp_out, N_NODES, 128, H_DIM);

    // 6) agg2 = spmm(gcn_out)  [50000 x 128]
    hipMemsetAsync(agg2, 0, NH2 * 4, stream);
    spmm_atomic<<<8192, 256, 0, stream>>>(e_src, e_dst, edge_w, gcn_out, agg2,
                                          N_EDGES, 128);

    // 7) mixture -> d_out (z_mean flat, then z_log_std flat)
    float* out_mean = (float*)d_out;
    float* out_ls   = out_mean + (long long)N_NODES * O_DIM;
    mix_kernel<<<(N_NODES * O_DIM + 255) / 256, 256, 0, stream>>>(
        agg2, mlp_out, mean_b, log_std_b, mix_w, out_mean, out_ls,
        N_NODES * O_DIM);
}
